// SALowFrequencyExtractor_66425964200570
// MI455X (gfx1250) — matrix operand order repaired and checked by reading, not compiled, not run on hardware
//
#include <hip/hip_runtime.h>
#include <hip/hip_bf16.h>
#include <math.h>

// ---------------------------------------------------------------------------
// Restormer-style block for MI455X (gfx1250, wave32, WMMA f16 16x16x32).
// Activations live in pixel-major f16 layout [pix][C] so every WMMA B-fragment
// is one contiguous v16h (32B) load; weights are pre-packed into WMMA
// A-fragment-major layout so every A-fragment is one contiguous v16h load.
// Inner GEMM bodies are branch-free with all fragment loads issued before the
// WMMA chain so the scheduler can overlap memory with matrix ops.
// ---------------------------------------------------------------------------

typedef __attribute__((ext_vector_type(16))) _Float16 v16h;
typedef __attribute__((ext_vector_type(8)))  float    v8f;

#define HW_    65536
#define W_     256
#define CDIM   96
#define BATCH  4
#define NPIX   (BATCH * HW_)   // 262144
#define HEADS  6

__device__ __forceinline__ v8f wmma_f16(v16h a, v16h b, v8f c) {
  return __builtin_amdgcn_wmma_f32_16x16x32_f16(
      /*neg_a=*/false, a, /*neg_b=*/false, b,
      /*c_mod=*/(short)0, c, /*reuse_a=*/false, /*reuse_b=*/false);
}

__device__ __forceinline__ float gelu_tanh(float x) {
  float u = 0.7978845608028654f * (x + 0.044715f * x * x * x);
  return 0.5f * x * (1.f + tanhf(u));
}

// ---------------------------------------------------------------------------
// Weight packing: [M][K] f32 -> WMMA A-fragment-major f16.
// Fragment layout (16-bit A 16x32): m = lane&15, kbase=(lane>>4)*16,
// element i -> K = kbase+i (contiguous, matches B loads).
// ---------------------------------------------------------------------------
__global__ void pack_pw(const float* __restrict__ W, _Float16* __restrict__ P,
                        int M, int K, int nfrag) {
  int idx  = blockIdx.x * blockDim.x + threadIdx.x;
  int frag = idx >> 5;
  if (frag >= nfrag) return;
  int lane  = idx & 31;
  int nkb   = K >> 5;
  int mt    = frag / nkb, kbt = frag % nkb;
  int m     = mt * 16 + (lane & 15);
  int kbase = (lane >> 4) * 16;
#pragma unroll
  for (int i = 0; i < 16; i++) {
    int kv = kbt * 32 + kbase + i;
    P[(long)idx * 16 + i] = (_Float16)W[m * K + kv];
  }
}

// qkv2 3x3 weights [288][288][3][3] -> fragment-major [mt(18)][kbt(9)][tap(9)][lane(32)][16]
__global__ void pack_conv(const float* __restrict__ W, _Float16* __restrict__ P) {
  int idx = blockIdx.x * blockDim.x + threadIdx.x;   // (mt*9+kbt)*9+tap times 32
  if (idx >= 18 * 9 * 9 * 32) return;
  int lane = idx & 31;
  int rest = idx >> 5;
  int tap  = rest % 9;
  int mk   = rest / 9;
  int kbt  = mk % 9, mt = mk / 9;
  int m    = mt * 16 + (lane & 15);
  int kbase = (lane >> 4) * 16;
#pragma unroll
  for (int i = 0; i < 16; i++) {
    int kv = kbt * 32 + kbase + i;
    P[(long)idx * 16 + i] = (_Float16)W[(m * 288 + kv) * 9 + tap];
  }
}

__global__ void zero_f32(float* __restrict__ p, int n) {
  int i = blockIdx.x * 256 + threadIdx.x;
  if (i < n) p[i] = 0.f;
}

// ---------------------------------------------------------------------------
// Channel LayerNorm: NCHW f32 -> pixel-major f16 [pix][96]
// ---------------------------------------------------------------------------
__global__ void ln_c_t(const float* __restrict__ X, const float* __restrict__ w,
                       const float* __restrict__ bi, _Float16* __restrict__ Y) {
  const long pix = (long)blockIdx.x * blockDim.x + threadIdx.x;
  const int  b   = (int)(pix >> 16);
  const int  hw  = (int)(pix & 65535);
  const float* xp = X + (((long)b * CDIM) << 16) + hw;
  float v[CDIM];
  float s = 0.f;
#pragma unroll
  for (int c = 0; c < CDIM; c++) { v[c] = xp[(long)c << 16]; s += v[c]; }
  float mu = s * (1.f / CDIM);
  float s2 = 0.f;
#pragma unroll
  for (int c = 0; c < CDIM; c++) { float d = v[c] - mu; s2 += d * d; }
  float rs = rsqrtf(s2 * (1.f / CDIM) + 1e-5f);
  _Float16* yp = Y + pix * CDIM;
#pragma unroll
  for (int c = 0; c < CDIM; c++)
    yp[c] = (_Float16)((v[c] - mu) * rs * w[c] + bi[c]);
}

// ---------------------------------------------------------------------------
// Pointwise-conv GEMM (f16 out): Y[pix][M] = Wpacked[M][K] @ X[pix][K]^T
// One wave per 16-pixel tile; TM m-tiles per wave reuse the B fragment.
// All fragment loads are issued before the WMMA chain (overlap-friendly).
// ---------------------------------------------------------------------------
template <int M, int K, int TM>
__global__ __launch_bounds__(32) void gemm_t_h(const _Float16* __restrict__ P,
                                               const _Float16* __restrict__ X,
                                               _Float16* __restrict__ Y) {
  constexpr int NKB = K / 32;
  const int lane = threadIdx.x;
  const int col  = lane & 15;
  const int half = lane >> 4;
  const long pix = (long)blockIdx.x * 16 + col;
  const int mt0  = blockIdx.y * TM;
  v8f zero = {};
  v8f acc[TM];
#pragma unroll
  for (int t = 0; t < TM; t++) acc[t] = zero;
  const _Float16* xrow = X + pix * K + half * 16;
  const _Float16* pbase = P + (((long)mt0 * NKB) * 32 + lane) * 16;
#pragma unroll
  for (int kbt = 0; kbt < NKB; kbt++) {
    v16h bf = *(const v16h*)(xrow + kbt * 32);
    v16h af[TM];
#pragma unroll
    for (int t = 0; t < TM; t++)
      af[t] = *(const v16h*)(pbase + ((long)(t * NKB + kbt) * 32) * 16);
#pragma unroll
    for (int t = 0; t < TM; t++)
      acc[t] = wmma_f16(af[t], bf, acc[t]);
  }
#pragma unroll
  for (int t = 0; t < TM; t++) {
    _Float16* yp = Y + pix * M + (mt0 + t) * 16 + half * 8;
#pragma unroll
    for (int r = 0; r < 8; r++) yp[r] = (_Float16)acc[t][r];
  }
}

// Same GEMM but adds NCHW f32 residual and writes NCHW f32 (96 channels).
template <int M, int K, int TM>
__global__ __launch_bounds__(32) void gemm_t_res_f32(const _Float16* __restrict__ P,
                                                     const _Float16* __restrict__ X,
                                                     const float* __restrict__ R,
                                                     float* __restrict__ Y) {
  constexpr int NKB = K / 32;
  const int lane = threadIdx.x;
  const int col  = lane & 15;
  const int half = lane >> 4;
  const long pix = (long)blockIdx.x * 16 + col;
  const int mt0  = blockIdx.y * TM;
  v8f zero = {};
  v8f acc[TM];
#pragma unroll
  for (int t = 0; t < TM; t++) acc[t] = zero;
  const _Float16* xrow = X + pix * K + half * 16;
  const _Float16* pbase = P + (((long)mt0 * NKB) * 32 + lane) * 16;
#pragma unroll
  for (int kbt = 0; kbt < NKB; kbt++) {
    v16h bf = *(const v16h*)(xrow + kbt * 32);
    v16h af[TM];
#pragma unroll
    for (int t = 0; t < TM; t++)
      af[t] = *(const v16h*)(pbase + ((long)(t * NKB + kbt) * 32) * 16);
#pragma unroll
    for (int t = 0; t < TM; t++)
      acc[t] = wmma_f16(af[t], bf, acc[t]);
  }
  const int b  = (int)(pix >> 16);
  const int hw = (int)(pix & 65535);
#pragma unroll
  for (int t = 0; t < TM; t++) {
#pragma unroll
    for (int r = 0; r < 8; r++) {
      int ch = (mt0 + t) * 16 + half * 8 + r;
      long idx = (((long)(b * CDIM + ch)) << 16) + hw;
      Y[idx] = R[idx] + acc[t][r];
    }
  }
}

// ---------------------------------------------------------------------------
// 3x3 conv 288->288, padding 1, implicit GEMM. 16-pixel row tiles, 9 taps.
// y-bound checks are wave-uniform (skip whole taps). Only x-edge tiles
// (x0==0 / x0==240) need per-lane predicates; interior tiles take a fully
// branch-free fast path so the scheduler can batch loads across the body.
// ---------------------------------------------------------------------------
__global__ __launch_bounds__(32) void conv3x3_gemm(const _Float16* __restrict__ P2,
                                                   const _Float16* __restrict__ X,
                                                   _Float16* __restrict__ Y) {
  const int lane = threadIdx.x;
  const int col  = lane & 15;
  const int half = lane >> 4;
  const long pix0 = (long)blockIdx.x * 16;
  const int b  = (int)(pix0 >> 16);
  const int y  = (int)((pix0 >> 8) & 255);
  const int x0 = (int)(pix0 & 255);
  const int mt0 = blockIdx.y * 6;
  v8f zero = {};
  v8f acc[6];
#pragma unroll
  for (int t = 0; t < 6; t++) acc[t] = zero;
  const bool xedge = (x0 == 0) || (x0 == 240);       // wave-uniform
  const _Float16* pbase = P2 + (((long)mt0 * 9) * 9 * 32 + lane) * 16;

  for (int dy = 0; dy < 3; dy++) {
    int yy = y + dy - 1;
    if (yy < 0 || yy > 255) continue;                // wave-uniform
    const long rowbase = ((long)b * 256 + yy) * 256;
    for (int dx = 0; dx < 3; dx++) {
      const int tap = dy * 3 + dx;
      if (!xedge) {
        // Fast path: branch-free, no EXEC churn.
        const _Float16* xb = X + (rowbase + x0 + col + dx - 1) * 288 + half * 16;
#pragma unroll
        for (int kbt = 0; kbt < 9; kbt++) {
          v16h bf = *(const v16h*)(xb + kbt * 32);
          v16h af[6];
#pragma unroll
          for (int t = 0; t < 6; t++)
            af[t] = *(const v16h*)(pbase + ((long)((t * 9 + kbt) * 9 + tap) * 32) * 16);
#pragma unroll
          for (int t = 0; t < 6; t++)
            acc[t] = wmma_f16(af[t], bf, acc[t]);
        }
      } else {
        // Edge path: per-lane x bounds predicate on the B load.
        int xx = x0 + col + dx - 1;
        bool ok = ((unsigned)xx) < 256u;
        const _Float16* xb = X + (rowbase + (ok ? xx : 0)) * 288 + half * 16;
        v16h hzero = {};
#pragma unroll
        for (int kbt = 0; kbt < 9; kbt++) {
          v16h bf = hzero;
          if (ok) bf = *(const v16h*)(xb + kbt * 32);
          v16h af[6];
#pragma unroll
          for (int t = 0; t < 6; t++)
            af[t] = *(const v16h*)(pbase + ((long)((t * 9 + kbt) * 9 + tap) * 32) * 16);
#pragma unroll
          for (int t = 0; t < 6; t++)
            acc[t] = wmma_f16(af[t], bf, acc[t]);
        }
      }
    }
  }
  const long pix = pix0 + col;
#pragma unroll
  for (int t = 0; t < 6; t++) {
    _Float16* yp = Y + pix * 288 + (mt0 + t) * 16 + half * 8;
#pragma unroll
    for (int r = 0; r < 8; r++) yp[r] = (_Float16)acc[t][r];
  }
}

// ---------------------------------------------------------------------------
// Sum of squares per (batch, q/k channel) -> SS[b][192]  (q:0..95, k:96..191)
// ---------------------------------------------------------------------------
__global__ void qk_sumsq(const _Float16* __restrict__ QKV2, float* __restrict__ SS) {
  int c = threadIdx.x;                 // 0..191
  int b = blockIdx.y;
  long p0 = (long)blockIdx.x * 512;
  float s = 0.f;
  for (int i = 0; i < 512; i++) {
    float v = (float)QKV2[(((long)b << 16) + p0 + i) * 288 + c];
    s += v * v;
  }
  atomicAdd(&SS[b * 192 + c], s);
}

// ---------------------------------------------------------------------------
// Gram matrix per (b,h): GR[bh][16][16] = q @ k^T over K = 65536 pixels.
// 8 waves split K, WMMA accumulate in registers, wave 0 sums the 8 partials via LDS.
// ---------------------------------------------------------------------------
__global__ void gram_qk(const _Float16* __restrict__ QKV2, float* __restrict__ GR) {
  __shared__ float red[8][32][8];
  const int bh = blockIdx.x;
  const int b = bh / HEADS, h = bh % HEADS;
  const int wave = threadIdx.x >> 5, lane = threadIdx.x & 31;
  const int col = lane & 15, half = lane >> 4;
  const long base = ((long)b) << 16;
  const int qo = h * 16, ko = 96 + h * 16;
  v8f acc = {};
  for (int kb = wave * 32; kb < HW_; kb += 8 * 32) {
    v16h af, bf;
#pragma unroll
    for (int i = 0; i < 16; i++) {
      int kpix = kb + half * 16 + i;
      af[i] = QKV2[(base + kpix) * 288 + qo + col];
    }
#pragma unroll
    for (int i = 0; i < 16; i++) {
      int kpix = kb + half * 16 + i;
      bf[i] = QKV2[(base + kpix) * 288 + ko + col];
    }
    acc = wmma_f16(af, bf, acc);
  }
#pragma unroll
  for (int r = 0; r < 8; r++) red[wave][lane][r] = acc[r];
  __syncthreads();
  if (wave == 0) {
#pragma unroll
    for (int r = 0; r < 8; r++) {
      float s = 0.f;
#pragma unroll
      for (int w2 = 0; w2 < 8; w2++) s += red[w2][lane][r];
      GR[bh * 256 + (r + half * 8) * 16 + col] = s;
    }
  }
}

// Fold 1/||q||, 1/||k||, head scale into Gram, then row softmax.
__global__ void attn_softmax(const float* __restrict__ GR, const float* __restrict__ SS,
                             const float* __restrict__ scale, float* __restrict__ AT) {
  int bh = blockIdx.x;
  int b = bh / HEADS, h = bh % HEADS;
  int c = threadIdx.x;  // row 0..15
  float nq = sqrtf(SS[b * 192 + h * 16 + c]);
  float sc = scale[h] / fmaxf(nq, 1e-12f);
  float row[16];
  float m = -1e30f;
#pragma unroll
  for (int d = 0; d < 16; d++) {
    float nk = sqrtf(SS[b * 192 + 96 + h * 16 + d]);
    float v = GR[bh * 256 + c * 16 + d] * sc / fmaxf(nk, 1e-12f);
    row[d] = v;
    m = fmaxf(m, v);
  }
  float s = 0.f;
#pragma unroll
  for (int d = 0; d < 16; d++) { row[d] = expf(row[d] - m); s += row[d]; }
  float inv = 1.f / s;
#pragma unroll
  for (int d = 0; d < 16; d++) AT[bh * 256 + c * 16 + d] = row[d] * inv;
}

// out[c,n] = sum_d attn[c,d] * v[d,n]  (rank-16, memory-bound -> VALU)
__global__ void attn_apply(const float* __restrict__ AT, const _Float16* __restrict__ QKV2,
                           _Float16* __restrict__ OUT) {
  __shared__ float a[256];
  const int h = blockIdx.y;
  const long pix = (long)blockIdx.x * 256 + threadIdx.x;
  const int b = (int)(pix >> 16);
  a[threadIdx.x] = AT[(b * HEADS + h) * 256 + threadIdx.x];
  __syncthreads();
  const _Float16* vp = QKV2 + pix * 288 + 192 + h * 16;
  float vv[16];
#pragma unroll
  for (int d = 0; d < 16; d++) vv[d] = (float)vp[d];
  _Float16* op = OUT + pix * CDIM + h * 16;
#pragma unroll
  for (int c = 0; c < 16; c++) {
    float s = 0.f;
#pragma unroll
    for (int d = 0; d < 16; d++) s += a[c * 16 + d] * vv[d];
    op[c] = (_Float16)s;
  }
}

// ---------------------------------------------------------------------------
// Grouped 3x3 conv (groups=192, 2ch/group) fused with gelu(x1)*x2 gate.
// Thread: one pixel, one channel pair. G[pix][192] f16 out.
// ---------------------------------------------------------------------------
__global__ void dwconv_gate(const _Float16* __restrict__ Hh, const float* __restrict__ DW,
                            _Float16* __restrict__ G) {
  __shared__ float w[72];
  const int gg = blockIdx.y;   // 0..95
  if (threadIdx.x < 72) {
    int t = threadIdx.x, o4 = t / 18, r = t % 18;
    int och = (o4 < 2) ? (2 * gg + o4) : (192 + 2 * gg + (o4 - 2));
    w[t] = DW[och * 18 + r];
  }
  __syncthreads();
  const long pix = (long)blockIdx.x * 256 + threadIdx.x;
  const int b = (int)(pix >> 16);
  const int y = (int)((pix >> 8) & 255);
  const int x = (int)(pix & 255);
  float a0 = 0.f, a1 = 0.f, a2 = 0.f, a3 = 0.f;
  for (int dy = 0; dy < 3; dy++) {
    int yy = y + dy - 1;
    if (yy < 0 || yy > 255) continue;
    for (int dx = 0; dx < 3; dx++) {
      int xx = x + dx - 1;
      if ((unsigned)xx >= 256u) continue;
      long pp = ((long)b * 256 + yy) * 256 + xx;
      const _Float16* hp = Hh + pp * 384;
      float i0 = (float)hp[2 * gg],       i1 = (float)hp[2 * gg + 1];
      float i2 = (float)hp[192 + 2 * gg], i3 = (float)hp[192 + 2 * gg + 1];
      int k = dy * 3 + dx;
      a0 += i0 * w[0 * 18 + k] + i1 * w[0 * 18 + 9 + k];
      a1 += i0 * w[1 * 18 + k] + i1 * w[1 * 18 + 9 + k];
      a2 += i2 * w[2 * 18 + k] + i3 * w[2 * 18 + 9 + k];
      a3 += i2 * w[3 * 18 + k] + i3 * w[3 * 18 + 9 + k];
    }
  }
  G[pix * 192 + 2 * gg]     = (_Float16)(gelu_tanh(a0) * a2);
  G[pix * 192 + 2 * gg + 1] = (_Float16)(gelu_tanh(a1) * a3);
}

// ---------------------------------------------------------------------------
extern "C" void kernel_launch(void* const* d_in, const int* in_sizes, int n_in,
                              void* d_out, int out_size, void* d_ws, size_t ws_size,
                              hipStream_t stream) {
  (void)in_sizes; (void)n_in; (void)out_size; (void)ws_size;
  const float* x     = (const float*)d_in[0];
  const float* ln1w  = (const float*)d_in[1];
  const float* ln1b  = (const float*)d_in[2];
  const float* qkv1w = (const float*)d_in[3];
  const float* qkv2w = (const float*)d_in[4];
  const float* projw = (const float*)d_in[5];
  const float* scale = (const float*)d_in[6];
  const float* ln2w  = (const float*)d_in[7];
  const float* ln2b  = (const float*)d_in[8];
  const float* pinw  = (const float*)d_in[9];
  const float* dww   = (const float*)d_in[10];
  const float* poutw = (const float*)d_in[11];
  float* out = (float*)d_out;

  char* ws = (char*)d_ws;
  size_t off = 0;
  auto take = [&](size_t bytes) -> char* {
    char* p = ws + off;
    off += (bytes + 255) & ~(size_t)255;
    return p;
  };
  _Float16* Pqkv1 = (_Float16*)take((size_t)288 * 96 * 2);
  _Float16* Pconv = (_Float16*)take((size_t)18 * 9 * 9 * 32 * 16 * 2);
  _Float16* Pproj = (_Float16*)take((size_t)96 * 96 * 2);
  _Float16* Ppin  = (_Float16*)take((size_t)384 * 96 * 2);
  _Float16* Ppout = (_Float16*)take((size_t)96 * 192 * 2);
  float* SS = (float*)take((size_t)4 * 192 * 4);
  float* GR = (float*)take((size_t)24 * 256 * 4);
  float* AT = (float*)take((size_t)24 * 256 * 4);
  char* region1 = take((size_t)NPIX * 96 * 2);    // xlnT -> attnoutT
  char* region2 = take((size_t)NPIX * 288 * 2);   // qkvT -> y2T
  char* region3 = take((size_t)NPIX * 288 * 2);   // qkv2T -> gT
  float* y1     = (float*)take((size_t)NPIX * 96 * 4);
  _Float16* hT  = (_Float16*)take((size_t)NPIX * 384 * 2);

  _Float16* xlnT  = (_Float16*)region1;
  _Float16* aoutT = (_Float16*)region1;   // alias (xlnT dead after qkv1 GEMM)
  _Float16* qkvT  = (_Float16*)region2;
  _Float16* y2T   = (_Float16*)region2;   // alias (qkvT dead after 3x3 conv)
  _Float16* qkv2T = (_Float16*)region3;
  _Float16* gT    = (_Float16*)region3;   // alias (qkv2T dead after attn_apply)

  // Weight packing + accumulator init (every call: graph-replay deterministic).
  pack_pw<<<(54 * 32 + 255) / 256, 256, 0, stream>>>(qkv1w, Pqkv1, 288, 96, 54);
  pack_conv<<<(18 * 9 * 9 * 32 + 255) / 256, 256, 0, stream>>>(qkv2w, Pconv);
  pack_pw<<<(18 * 32 + 255) / 256, 256, 0, stream>>>(projw, Pproj, 96, 96, 18);
  pack_pw<<<(72 * 32 + 255) / 256, 256, 0, stream>>>(pinw, Ppin, 384, 96, 72);
  pack_pw<<<(36 * 32 + 255) / 256, 256, 0, stream>>>(poutw, Ppout, 96, 192, 36);
  zero_f32<<<(768 + 255) / 256, 256, 0, stream>>>(SS, 768);

  // --- Attention branch ---
  ln_c_t<<<NPIX / 256, 256, 0, stream>>>(x, ln1w, ln1b, xlnT);
  gemm_t_h<288, 96, 6><<<dim3(NPIX / 16, 3), 32, 0, stream>>>(Pqkv1, xlnT, qkvT);
  conv3x3_gemm<<<dim3(NPIX / 16, 3), 32, 0, stream>>>(Pconv, qkvT, qkv2T);
  qk_sumsq<<<dim3(HW_ / 512, BATCH), 192, 0, stream>>>(qkv2T, SS);
  gram_qk<<<BATCH * HEADS, 256, 0, stream>>>(qkv2T, GR);
  attn_softmax<<<BATCH * HEADS, 16, 0, stream>>>(GR, SS, scale, AT);
  attn_apply<<<dim3(NPIX / 256, HEADS), 256, 0, stream>>>(AT, qkv2T, aoutT);
  gemm_t_res_f32<96, 96, 6><<<dim3(NPIX / 16, 1), 32, 0, stream>>>(Pproj, aoutT, x, y1);

  // --- MLP branch ---
  ln_c_t<<<NPIX / 256, 256, 0, stream>>>(y1, ln2w, ln2b, y2T);
  gemm_t_h<384, 96, 6><<<dim3(NPIX / 16, 4), 32, 0, stream>>>(Ppin, y2T, hT);
  dwconv_gate<<<dim3(NPIX / 256, 96), 256, 0, stream>>>(hT, dww, gT);
  gemm_t_res_f32<96, 192, 6><<<dim3(NPIX / 16, 1), 32, 0, stream>>>(Ppout, gT, y1, out);
}
